// BilateralConvFlex_68418829025451
// MI455X (gfx1250) — matrix operand order repaired and checked by reading, not compiled, not run on hardware
//
#include <hip/hip_runtime.h>

// Problem constants (match reference)
#define B_    4
#define CIN   64
#define NIN   8192
#define NOUT  8192
#define D1    4
#define FF    15
#define HH    16384
#define HP1   (HH + 1)
#define CMID  64
#define COUT  64
#define K0    (FF * CIN)      // 960
#define KS0   (K0 / 32)       // 30 k-steps for GEMM1
#define KS1   (CMID / 32)     // 2  k-steps for GEMM2
#define LEAK  0.01f

#define HT    32              // vertex tile per workgroup
#define SP    968             // padded spread row (960 + 8) -> conflict-free ds_read_b128
#define HPAD  72              // padded h0 row (64 + 8)

typedef __bf16 bf16;
typedef __attribute__((ext_vector_type(16))) __bf16 v16bf;
typedef __attribute__((ext_vector_type(8)))  float  v8f;

// Async direct-to-LDS staging via inline asm (ASYNCcnt path, §15.18.3 op 98).
#if defined(__gfx1250__)
#define USE_ASYNC_LDS 1
#else
#define USE_ASYNC_LDS 0
#endif

union V16U { uint4 q[2]; v16bf v; };

// ---------------------------------------------------------------- zero scratch
__global__ void k_zero(float* __restrict__ p, long long n) {
  long long i = (long long)blockIdx.x * blockDim.x + threadIdx.x;
  long long s = (long long)gridDim.x * blockDim.x;
  for (; i < n; i += s) p[i] = 0.0f;
}

// ---------------------------------------------------------------- splat (scatter)
__global__ void k_splat(const float* __restrict__ feat,   // (B,C,N)
                        const float* __restrict__ bary,   // (B,D1,N)
                        const int*   __restrict__ loff,   // (B,D1,N) global offsets
                        float* __restrict__ splat,        // (B*HP1, C)
                        float* __restrict__ ones)         // (B*HP1)
{
  int e = blockIdx.x * blockDim.x + threadIdx.x;          // over B*D1*N, n fastest
  if (e >= B_ * D1 * NIN) return;
  int n = e % NIN;
  int d = (e / NIN) % D1;
  int b = e / (NIN * D1);
  float w = bary[((size_t)b * D1 + d) * NIN + n];
  int   v = loff[((size_t)b * D1 + d) * NIN + n] + 1;     // already includes b*(H+1) base
  atomicAdd(&ones[v], w);
  const float* f = feat + (size_t)b * CIN * NIN + n;
  float*       s = splat + (size_t)v * CIN;
  #pragma unroll 4
  for (int c = 0; c < CIN; ++c)
    atomicAdd(&s[c], w * f[(size_t)c * NIN]);
}

// ---------------------------------------------------------------- normalize -> bf16 table
__global__ void k_norm(const float* __restrict__ splat,
                       const float* __restrict__ ones,
                       bf16* __restrict__ sb)
{
  long long i = (long long)blockIdx.x * blockDim.x + threadIdx.x;
  long long n = (long long)B_ * HP1 * CIN;
  if (i >= n) return;
  float denom = ones[i >> 6] + 1e-5f;
  sb[i] = (bf16)(splat[i] / denom);
}

// ---------------------------------------------------------------- pack W0/W1 into WMMA A-fragment order
// 16-bit A layout (16x32): lane<16 holds M=lane, K slots {0..7,16..23};
// lane>=16 holds M=lane-16, K slots {8..15,24..31}.  K index = f*64 + c.
__global__ void k_pack(const float* __restrict__ W0,      // (CMID,CIN,FF)
                       const float* __restrict__ W1,      // (COUT,CMID)
                       bf16* __restrict__ w0p, bf16* __restrict__ w1p)
{
  int i  = blockIdx.x * blockDim.x + threadIdx.x;
  int n0 = 4 * KS0 * 32 * 16;
  int n1 = 4 * KS1 * 32 * 16;
  if (i < n0) {
    int s    = i & 15;
    int lane = (i >> 4) & 31;
    int ks   = (i >> 9) % KS0;
    int ot   = (i >> 9) / KS0;
    int m    = ot * 16 + (lane & 15);
    int koff = (s < 8 ? s : s + 8) + (lane < 16 ? 0 : 8);
    int k    = ks * 32 + koff;
    int f = k >> 6, c = k & 63;
    w0p[i] = (bf16)W0[((size_t)m * CIN + c) * FF + f];
  } else if (i < n0 + n1) {
    int j    = i - n0;
    int s    = j & 15;
    int lane = (j >> 4) & 31;
    int ks   = (j >> 9) % KS1;
    int pt   = (j >> 9) / KS1;
    int m    = pt * 16 + (lane & 15);
    int koff = (s < 8 ? s : s + 8) + (lane < 16 ? 0 : 8);
    w1p[j] = (bf16)W1[(size_t)m * CMID + ks * 32 + koff];
  }
}

// ---------------------------------------------------------------- blur: gather + two WMMA GEMMs
__global__ void __launch_bounds__(256)
k_blur(const bf16* __restrict__ sb,        // (B*HP1, 64) normalized bf16 splat
       const int*  __restrict__ nbr,       // (B, FF, H), values in [-1, H)
       const bf16* __restrict__ w0p,
       const bf16* __restrict__ w1p,
       const float* __restrict__ b0,
       const float* __restrict__ b1,
       float* __restrict__ blurred)        // (B, H, 64) vertex-major
{
  __shared__ __align__(16) bf16 sSpread[HT][SP];          // 61,952 B
  // h0 exchange buffer overlaid on sSpread (spread is dead by then)
  bf16 (*sH0)[HPAD] = reinterpret_cast<bf16 (*)[HPAD]>(&sSpread[0][0]);

  const int tid   = threadIdx.x;
  const int b     = blockIdx.x / (HH / HT);
  const int hbase = (blockIdx.x % (HH / HT)) * HT;

  // ---- stage spread: 15 neighbors x 32 vertices x 64 ch, 16-byte units ----
  // Async direct-to-LDS copies (ASYNCcnt): no VGPR round-trip for the 60KB tile.
  for (int i = tid; i < FF * HT * 8; i += 256) {          // 3840 units
    int ch = i & 7;                                       // 8 bf16 chunk
    int h  = (i >> 3) & (HT - 1);
    int f  = i >> 8;                                      // HT*8 == 256
    int nb = nbr[((size_t)b * FF + f) * HH + hbase + h] + 1;     // [0, H]
    size_t row = (size_t)b * HP1 + nb;
    const bf16* src = sb + row * CIN + ch * 8;
    bf16*       dst = &sSpread[h][f * CIN + ch * 8];
#if USE_ASYNC_LDS
    // GV mode: vdst = per-lane LDS byte address (low 32 bits of flat shared ptr),
    // vaddr = 64-bit global address.  Tracked by ASYNCcnt.
    unsigned           ldsa = (unsigned)(uintptr_t)dst;
    unsigned long long ga   = (unsigned long long)(uintptr_t)src;
    asm volatile("global_load_async_to_lds_b128 %0, %1, off"
                 :: "v"(ldsa), "v"(ga) : "memory");
#else
    *reinterpret_cast<uint4*>(dst) = *reinterpret_cast<const uint4*>(src);
#endif
  }
#if USE_ASYNC_LDS
  asm volatile("s_wait_asynccnt 0" ::: "memory");
#endif
  __syncthreads();

  const int wave  = tid >> 5;
  const int lane  = tid & 31;
  const int ot    = wave & 3;                 // 16-row output tile (o / p)
  const int hs    = (wave >> 2) * 16;         // 16-vertex sub-tile
  const int hl    = hs + (lane & 15);         // this lane's vertex (B/C column N)
  const int khalf = (lane < 16) ? 0 : 8;      // B-fragment K half per ISA layout
  const int mo    = (lane < 16) ? 0 : 8;      // C/D row offset per ISA layout

  // ---- GEMM1: h0[o,h] = sum_k W0[o,k] * spread[k,h], K = 960 ----
  v8f acc = {};
  const v16bf* A1 = reinterpret_cast<const v16bf*>(w0p) + (ot * KS0) * 32 + lane;
  #pragma unroll 2
  for (int ks = 0; ks < KS0; ++ks) {
    v16bf a = A1[ks * 32];
    V16U bu;
    const bf16* brow = &sSpread[hl][ks * 32 + khalf];
    bu.q[0] = *reinterpret_cast<const uint4*>(brow);
    bu.q[1] = *reinterpret_cast<const uint4*>(brow + 16);
    acc = __builtin_amdgcn_wmma_f32_16x16x32_bf16(false, a, false, bu.v,
                                                  (short)0, acc, false, false);
  }
  __syncthreads();   // spread buffer dead everywhere before h0 overlay writes

  // bias + LeakyReLU, deposit h0 (bf16) for the cross-wave second GEMM
  #pragma unroll
  for (int r = 0; r < 8; ++r) {
    int o = ot * 16 + r + mo;
    float v = acc[r] + b0[o];
    v = (v > 0.0f) ? v : LEAK * v;
    sH0[hl][o] = (bf16)v;
  }
  __syncthreads();

  // ---- GEMM2: out[p,h] = sum_o W1[p,o] * h0[o,h], K = 64 ----
  v8f acc2 = {};
  const v16bf* A2 = reinterpret_cast<const v16bf*>(w1p) + (ot * KS1) * 32 + lane;
  #pragma unroll
  for (int ks = 0; ks < KS1; ++ks) {
    v16bf a = A2[ks * 32];
    V16U bu;
    const bf16* brow = &sH0[hl][ks * 32 + khalf];
    bu.q[0] = *reinterpret_cast<const uint4*>(brow);
    bu.q[1] = *reinterpret_cast<const uint4*>(brow + 16);
    acc2 = __builtin_amdgcn_wmma_f32_16x16x32_bf16(false, a, false, bu.v,
                                                   (short)0, acc2, false, false);
  }

  // bias + store: each lane owns 8 consecutive channels of one vertex
  {
    int pb = ot * 16 + mo;
    float4 v0, v1;
    v0.x = acc2[0] + b1[pb + 0]; v0.y = acc2[1] + b1[pb + 1];
    v0.z = acc2[2] + b1[pb + 2]; v0.w = acc2[3] + b1[pb + 3];
    v1.x = acc2[4] + b1[pb + 4]; v1.y = acc2[5] + b1[pb + 5];
    v1.z = acc2[6] + b1[pb + 6]; v1.w = acc2[7] + b1[pb + 7];
    float* dst = blurred + ((size_t)b * HH + hbase + hl) * COUT + pb;
    *reinterpret_cast<float4*>(dst)     = v0;
    *reinterpret_cast<float4*>(dst + 4) = v1;
  }
}

// ---------------------------------------------------------------- slice (gather + transpose)
__global__ void __launch_bounds__(256)
k_slice(const float* __restrict__ blurred,  // (B, H, 64)
        const float* __restrict__ obary,    // (B, D1, N_out)
        const int*   __restrict__ ooff,     // (B, D1, N_out), values in [0, H)
        const float* __restrict__ bias,
        float* __restrict__ out)            // (B, C_out, N_out)
{
  __shared__ float sOut[16][COUT + 1];
  int b     = blockIdx.x / (NOUT / 16);
  int nbase = (blockIdx.x % (NOUT / 16)) * 16;
  int tid = threadIdx.x;
  int pt  = tid >> 4;                       // point within tile
  int cq  = tid & 15;                       // channel quad
  int n   = nbase + pt;

  float4 acc = {0.f, 0.f, 0.f, 0.f};
  #pragma unroll
  for (int d = 0; d < D1; ++d) {
    float w = obary[((size_t)b * D1 + d) * NOUT + n];
    int   v = ooff [((size_t)b * D1 + d) * NOUT + n];
    float4 t = *reinterpret_cast<const float4*>(
        blurred + ((size_t)b * HH + v) * COUT + cq * 4);
    acc.x += w * t.x; acc.y += w * t.y; acc.z += w * t.z; acc.w += w * t.w;
  }
  acc.x += bias[cq * 4 + 0]; acc.y += bias[cq * 4 + 1];
  acc.z += bias[cq * 4 + 2]; acc.w += bias[cq * 4 + 3];
  sOut[pt][cq * 4 + 0] = acc.x; sOut[pt][cq * 4 + 1] = acc.y;
  sOut[pt][cq * 4 + 2] = acc.z; sOut[pt][cq * 4 + 3] = acc.w;
  __syncthreads();
  for (int i = tid; i < 16 * COUT; i += 256) {
    int p = i >> 4, no = i & 15;
    out[((size_t)b * COUT + p) * NOUT + nbase + no] = sOut[no][p];
  }
}

// ---------------------------------------------------------------- host launcher
extern "C" void kernel_launch(void* const* d_in, const int* in_sizes, int n_in,
                              void* d_out, int out_size, void* d_ws, size_t ws_size,
                              hipStream_t stream)
{
  const float* features = (const float*)d_in[0];
  const float* in_bary  = (const float*)d_in[1];
  const int*   in_off   = (const int*)  d_in[2];
  const int*   nbr      = (const int*)  d_in[3];
  const float* out_bary = (const float*)d_in[4];
  const int*   out_off  = (const int*)  d_in[5];
  const float* W0       = (const float*)d_in[6];
  const float* b0       = (const float*)d_in[7];
  const float* W1       = (const float*)d_in[8];
  const float* b1       = (const float*)d_in[9];
  const float* bias     = (const float*)d_in[10];
  (void)in_sizes; (void)n_in; (void)out_size; (void)ws_size;

  char*  ws  = (char*)d_ws;
  size_t off = 0;
  auto take = [&](size_t bytes) {
    char* p = ws + off;
    off = (off + bytes + 255) & ~(size_t)255;
    return p;
  };
  float* splat   = (float*)take((size_t)B_ * HP1 * CIN * sizeof(float));
  float* ones    = (float*)take((size_t)B_ * HP1 * sizeof(float));
  bf16*  sb      = (bf16*) take((size_t)B_ * HP1 * CIN * sizeof(bf16));
  bf16*  w0p     = (bf16*) take((size_t)4 * KS0 * 32 * 16 * sizeof(bf16));
  bf16*  w1p     = (bf16*) take((size_t)4 * KS1 * 32 * 16 * sizeof(bf16));
  float* blurred = (float*)take((size_t)B_ * HH * COUT * sizeof(float));

  k_zero<<<2048, 256, 0, stream>>>(splat, (long long)B_ * HP1 * CIN);
  k_zero<<<64,   256, 0, stream>>>(ones,  (long long)B_ * HP1);

  k_splat<<<(B_ * D1 * NIN) / 256, 256, 0, stream>>>(features, in_bary, in_off,
                                                     splat, ones);

  long long nn = (long long)B_ * HP1 * CIN;
  k_norm<<<(int)((nn + 255) / 256), 256, 0, stream>>>(splat, ones, sb);

  int npack = 4 * KS0 * 32 * 16 + 4 * KS1 * 32 * 16;
  k_pack<<<(npack + 255) / 256, 256, 0, stream>>>(W0, W1, w0p, w1p);

  k_blur<<<B_ * (HH / HT), 256, 0, stream>>>(sb, nbr, w0p, w1p, b0, b1, blurred);

  k_slice<<<B_ * (NOUT / 16), 256, 0, stream>>>(blurred, out_bary, out_off, bias,
                                                (float*)d_out);
}
